// AttentionModel_19121194402186
// MI455X (gfx1250) — matrix operand order repaired and checked by reading, not compiled, not run on hardware
//
#include <hip/hip_runtime.h>

#define N_RULES 128
#define D_L     128
#define NHIDDEN 256
#define NCLASS  10
#define NFEAT   256

typedef __attribute__((ext_vector_type(16))) __bf16 v16bf;
typedef __attribute__((ext_vector_type(8)))  __bf16 v8bf;
typedef __attribute__((ext_vector_type(8)))  float  v8f;

__device__ __forceinline__ unsigned short f2bf_rne(float f) {
  unsigned int u = __float_as_uint(f);
  unsigned int r = u + 0x7FFFu + ((u >> 16) & 1u);
  return (unsigned short)(r >> 16);
}

// W1 (NFEAT x NHIDDEN) -> W1T bf16 (NHIDDEN x NFEAT); W2 (NHIDDEN x N_RULES) -> W2T bf16 (N_RULES x NHIDDEN)
__global__ void cvt_weights_kernel(const float* __restrict__ W1, const float* __restrict__ W2,
                                   unsigned short* __restrict__ W1T, unsigned short* __restrict__ W2T) {
  int tid = blockIdx.x * blockDim.x + threadIdx.x;
  int stride = gridDim.x * blockDim.x;
  for (int i = tid; i < NFEAT * NHIDDEN; i += stride) {
    int k = i >> 8;            // feature (row of W1)
    int n = i & 255;           // hidden column
    W1T[n * NFEAT + k] = f2bf_rne(W1[i]);
  }
  for (int i = tid; i < NHIDDEN * N_RULES; i += stride) {
    int k = i >> 7;            // hidden
    int c = i & 127;           // rule
    W2T[c * NHIDDEN + k] = f2bf_rne(W2[i]);
  }
}

__global__ __launch_bounds__(256)
void attn_kernel(const float* __restrict__ x_lf, const float* __restrict__ x_l,
                 const float* __restrict__ b1, const float* __restrict__ b2,
                 const unsigned short* __restrict__ W1Tu, const unsigned short* __restrict__ W2Tu,
                 float* __restrict__ out_y, float* __restrict__ cov) {
  const __bf16* W1T = (const __bf16*)W1Tu;
  const __bf16* W2T = (const __bf16*)W2Tu;

  // per-wave 16x128 bf16 h staging, row stride 136 (272B: 16B-aligned, bank-phase shifted)
  __shared__ __bf16 h_lds[8 * 16 * 136];

  const int lane = threadIdx.x & 31;
  const int wave = threadIdx.x >> 5;
  const int lcol = lane & 15;
  const int lhi  = lane >> 4;                 // 0: rows/K low half, 1: high half
  const int rowBase = blockIdx.x * 128 + wave * 16;
  __bf16* hw = h_lds + wave * (16 * 136);

  v8f acc2[8] = {};                           // z accumulators: 8 tiles x 16 cols = 128 rules

  #pragma unroll
  for (int p = 0; p < 2; ++p) {               // hidden dimension processed in two 128-col chunks
    const int colBase = p * 128;
    v8f acc1[8] = {};

    // ---- GEMM1: h_chunk = x @ W1[:, colBase:colBase+128] ----
    const int arow = rowBase + lcol;          // A layout: M = lane%16 for both halves
    #pragma unroll
    for (int kb = 0; kb < 8; ++kb) {          // K = 256 in steps of 32
      int kc = kb * 32 + (lhi ? 16 : 0);      // first K-run start (feature space)
      const float* src = (kc < N_RULES)
                       ? (x_lf + (size_t)arow * N_RULES + kc)
                       : (x_l  + (size_t)arow * D_L + (kc - N_RULES));
      float4 f0 = ((const float4*)src)[0];
      float4 f1 = ((const float4*)src)[1];
      float4 f2 = ((const float4*)(src + 8))[0];
      float4 f3 = ((const float4*)(src + 8))[1];
      v16bf a;
      a[0]=(__bf16)f0.x;  a[1]=(__bf16)f0.y;  a[2]=(__bf16)f0.z;  a[3]=(__bf16)f0.w;
      a[4]=(__bf16)f1.x;  a[5]=(__bf16)f1.y;  a[6]=(__bf16)f1.z;  a[7]=(__bf16)f1.w;
      a[8]=(__bf16)f2.x;  a[9]=(__bf16)f2.y;  a[10]=(__bf16)f2.z; a[11]=(__bf16)f2.w;
      a[12]=(__bf16)f3.x; a[13]=(__bf16)f3.y; a[14]=(__bf16)f3.z; a[15]=(__bf16)f3.w;

      int kg = kb * 32 + (lhi ? 16 : 0);      // B layout: K lo/hi half by lane half
      #pragma unroll
      for (int t = 0; t < 8; ++t) {
        int n = colBase + t * 16 + lcol;      // hidden column = lane%16 within tile
        v16bf bfr = *(const v16bf*)(W1T + (size_t)n * NFEAT + kg);
        acc1[t] = __builtin_amdgcn_wmma_f32_16x16x32_bf16(false, a, false, bfr,
                                                          (short)0, acc1[t], false, false);
      }
    }

    // ---- bias + tanh, stage bf16 h to wave-private LDS ----
    #pragma unroll
    for (int t = 0; t < 8; ++t) {
      float bias = b1[colBase + t * 16 + lcol];
      #pragma unroll
      for (int v = 0; v < 8; ++v) {
        int r = v + lhi * 8;                  // C layout: row = vgpr + 8*laneHalf
        float hv = tanhf(acc1[t][v] + bias);
        hw[r * 136 + t * 16 + lcol] = (__bf16)hv;
      }
    }

    // ---- GEMM2 partial: z += h_chunk @ W2[colBase:colBase+128, :] ----
    #pragma unroll
    for (int kb = 0; kb < 4; ++kb) {          // 128 local K in steps of 32
      int kl = kb * 32 + (lhi ? 16 : 0);
      const __bf16* hsrc = hw + lcol * 136 + kl;
      v8bf h0 = *(const v8bf*)(hsrc);
      v8bf h1 = *(const v8bf*)(hsrc + 8);
      v16bf a;
      #pragma unroll
      for (int i = 0; i < 8; ++i) { a[i] = h0[i]; a[i + 8] = h1[i]; }

      int kg = colBase + kb * 32 + (lhi ? 16 : 0);
      #pragma unroll
      for (int t = 0; t < 8; ++t) {
        int c = t * 16 + lcol;                // rule column
        v16bf bfr = *(const v16bf*)(W2T + (size_t)c * NHIDDEN + kg);
        acc2[t] = __builtin_amdgcn_wmma_f32_16x16x32_bf16(false, a, false, bfr,
                                                          (short)0, acc2[t], false, false);
      }
    }
  }

  // ---- fused epilogue: softmax(128), coverage, class buckets, log-softmax(10) ----
  #pragma unroll
  for (int v = 0; v < 8; ++v) {
    int row = rowBase + lhi * 8 + v;
    const float* lfrow = x_lf + (size_t)row * N_RULES;
    float z[8], vote[8];
    #pragma unroll
    for (int t = 0; t < 8; ++t) {
      z[t]    = acc2[t][v] + b2[t * 16 + lcol];
      vote[t] = lfrow[t * 16 + lcol];
    }
    float m = z[0];
    #pragma unroll
    for (int t = 1; t < 8; ++t) m = fmaxf(m, z[t]);
    #pragma unroll
    for (int s = 1; s < 16; s <<= 1) m = fmaxf(m, __shfl_xor(m, s, 16));
    float ssum = 0.f, sc[8];
    #pragma unroll
    for (int t = 0; t < 8; ++t) { sc[t] = __expf(z[t] - m); ssum += sc[t]; }
    #pragma unroll
    for (int s = 1; s < 16; s <<= 1) ssum += __shfl_xor(ssum, s, 16);
    float inv = 1.0f / ssum;
    #pragma unroll
    for (int t = 0; t < 8; ++t) {
      sc[t] *= inv;
      cov[(size_t)row * N_RULES + t * 16 + lcol] = (vote[t] >= 0.f) ? sc[t] : 0.f;
    }
    float buck[NCLASS];
    #pragma unroll
    for (int c = 0; c < NCLASS; ++c) {
      float b = 0.f;
      #pragma unroll
      for (int t = 0; t < 8; ++t)
        b += (vote[t] >= 0.f && (int)vote[t] == c) ? sc[t] : 0.f;
      #pragma unroll
      for (int s = 1; s < 16; s <<= 1) b += __shfl_xor(b, s, 16);
      buck[c] = b;
    }
    float bm = buck[0];
    #pragma unroll
    for (int c = 1; c < NCLASS; ++c) bm = fmaxf(bm, buck[c]);
    float bs = 0.f;
    #pragma unroll
    for (int c = 0; c < NCLASS; ++c) bs += __expf(buck[c] - bm);
    float lse = __logf(bs);
    if (lcol == 0) {
      #pragma unroll
      for (int c = 0; c < NCLASS; ++c)
        out_y[(size_t)row * NCLASS + c] = buck[c] - bm - lse;
    }
  }
}

extern "C" void kernel_launch(void* const* d_in, const int* in_sizes, int n_in,
                              void* d_out, int out_size, void* d_ws, size_t ws_size,
                              hipStream_t stream) {
  const float* x_lf = (const float*)d_in[0];
  const float* x_l  = (const float*)d_in[1];
  const float* W1   = (const float*)d_in[2];
  const float* b1   = (const float*)d_in[3];
  const float* W2   = (const float*)d_in[4];
  const float* b2   = (const float*)d_in[5];

  int N = in_sizes[0] / N_RULES;              // 262144

  unsigned short* W1T = (unsigned short*)d_ws;                       // 128 KB bf16
  unsigned short* W2T = W1T + (size_t)NFEAT * NHIDDEN;               // +64 KB bf16

  float* out_y = (float*)d_out;               // (N, 10) log-softmax
  float* cov   = out_y + (size_t)N * NCLASS;  // (N, 128) coverage scores

  cvt_weights_kernel<<<128, 256, 0, stream>>>(W1, W2, W1T, W2T);
  attn_kernel<<<N / 128, 256, 0, stream>>>(x_lf, x_l, b1, b2, W1T, W2T, out_y, cov);
}
